// Rot_Inv_ConvNet_18150531793364
// MI455X (gfx1250) — matrix operand order repaired and checked by reading, not compiled, not run on hardware
//
#include <hip/hip_runtime.h>

typedef __bf16 bf16_t;
typedef __attribute__((ext_vector_type(8)))  __bf16 v8bf;
typedef __attribute__((ext_vector_type(16))) __bf16 v16bf;
typedef __attribute__((ext_vector_type(8)))  float  v8f;

struct __align__(16) Tap { int idx[4]; float w[4]; };  // 4 gather indices + 4 bilinear weights

static __device__ inline v16bf cat16(v8bf lo, v8bf hi) {
  return __builtin_shufflevector(lo, hi, 0,1,2,3,4,5,6,7,8,9,10,11,12,13,14,15);
}

// ---------------- tap table: 4-point gather metadata per (h,w,tap) ----------------
__global__ void tapprep(const float* __restrict__ coords, Tap* __restrict__ taps, int H, int W) {
  int t = blockIdx.x * blockDim.x + threadIdx.x;
  int HW = H * W;
  if (t >= HW * 9) return;
  int s = t / 9, tap = t - s * 9;
  int h = s / W, w = s - h * W;
  int kh = tap / 3, kw = tap - kh * 3;
  float py = (float)(h - 1 + kh) + coords[s * 18 + 2 * tap];
  float px = (float)(w - 1 + kw) + coords[s * 18 + 2 * tap + 1];
  float y0 = floorf(py), x0 = floorf(px);
  float wy1 = py - y0, wy0 = 1.f - wy1;
  float wx1 = px - x0, wx0 = 1.f - wx1;
  float ys[2] = {y0, y0 + 1.f}, xs[2] = {x0, x0 + 1.f};
  float wy[2] = {wy0, wy1},     wx[2] = {wx0, wx1};
  Tap tp;
#pragma unroll
  for (int j = 0; j < 4; ++j) {
    float yf = ys[j >> 1], xf = xs[j & 1];
    bool valid = (yf >= 0.f) && (yf < (float)H) && (xf >= 0.f) && (xf < (float)W);
    int yi = min(max((int)yf, 0), H - 1);
    int xi = min(max((int)xf, 0), W - 1);
    tp.idx[j] = yi * W + xi;
    tp.w[j]   = valid ? wy[j >> 1] * wx[j & 1] : 0.f;
  }
  taps[t] = tp;
}

// ---------------- fp32 -> bf16 ----------------
__global__ void f2bf(const float* __restrict__ in, bf16_t* __restrict__ out, long n) {
  long i = (long)blockIdx.x * blockDim.x + threadIdx.x;
  if (i < n) out[i] = (bf16_t)in[i];
}

// ---------------- weight prep: [Cout,Cin,3,3] fp32 -> [Cout, Kpad] bf16 ----------------
__global__ void wprep(const float* __restrict__ w, bf16_t* __restrict__ wt, int cout, int k, int kp) {
  int i = blockIdx.x * blockDim.x + threadIdx.x;
  if (i >= cout * kp) return;
  int o = i / kp, kk = i - o * kp;
  wt[i] = (kk < k) ? (bf16_t)w[o * k + kk] : (bf16_t)0.f;
}

// ---------------- fused deform-im2col (+BN+ReLU of producer) + WMMA GEMM + BN partial stats ----
// act: [B,CIN,H,W] bf16 (raw conv if BN_IN, else already activated)
// wt:  [COUT,KP] bf16    out: [B,COUT,H,W] bf16 RAW conv result
// ps1/ps2: [COUT, gridDim.x] deterministic per-block partial sums / sum-of-squares
template <int CIN, int COUT, int H, int W, int WGM, int WGN, bool BN_IN>
__global__ __launch_bounds__(256) void dconv_gemm(const bf16_t* __restrict__ act,
                                                  const bf16_t* __restrict__ wt,
                                                  const Tap* __restrict__ taps,
                                                  const float* __restrict__ scshIn,
                                                  bf16_t* __restrict__ out,
                                                  float* __restrict__ ps1,
                                                  float* __restrict__ ps2) {
  constexpr int HW = H * W;
  constexpr int LOG_HW = __builtin_ctz(HW);
  constexpr int K  = CIN * 9;
  constexpr int KP = (K + 31) & ~31;
  constexpr int KT = KP / 32;
  constexpr int BM = 128;
  constexpr int BN = COUT;
  constexpr int TM = BM / (WGM * 16);
  constexpr int TN = BN / (WGN * 16);
  constexpr int LDA = 40;  // +8 bf16 pad to spread LDS banks

  __shared__ bf16_t sA[2][BM * LDA];     // double-buffered A tile
  __shared__ float  sRed[2][8][BN];      // BN-stat cross-wave reduction

  const int tid    = threadIdx.x;
  const int lane   = tid & 31;
  const int waveId = tid >> 5;
  const int wm = waveId % WGM, wn = waveId / WGM;
  const int wmBase = wm * (TM * 16);
  const int wnBase = wn * (TN * 16);
  const long mBase = (long)blockIdx.x * BM;

  const int bcol  = tid & 31;   // builder: fixed column
  const int brow0 = tid >> 5;   // builder: first row (stride 8)

  // fragment addressing (ISA 7.12.2)
  const int arow = lane & 15;
  const int akb  = (lane >> 4) * 8;
  const int bnc  = lane & 15;
  const int bkb  = (lane >> 4) * 8;

  v8f acc[TM][TN];
#pragma unroll
  for (int i = 0; i < TM; ++i)
#pragma unroll
    for (int j = 0; j < TN; ++j) acc[i][j] = (v8f)(0.f);

  auto build = [&](int kt, bf16_t* dst) {
    const int col = kt * 32 + bcol;
    const bool cv = (col < K);
    const int c   = cv ? (col / 9) : 0;
    const int tap = cv ? (col - (col / 9) * 9) : 0;
    float sc = 1.f, sh = 0.f;
    if (BN_IN && cv) { sc = scshIn[c]; sh = scshIn[CIN + c]; }
#pragma unroll 4
    for (int i = 0; i < BM / 8; ++i) {
      int row = brow0 + i * 8;
      long m = mBase + row;
      int b = (int)(m >> LOG_HW);
      int s = (int)(m & (HW - 1));
      float v = 0.f;
      if (cv) {
        Tap t = taps[s * 9 + tap];
        const bf16_t* plane = act + ((long)b * CIN + c) * HW;
        if (BN_IN) {  // producer BN+ReLU applied per gathered corner (linear interp after pointwise op)
          v = t.w[0] * fmaxf(sc * (float)plane[t.idx[0]] + sh, 0.f) +
              t.w[1] * fmaxf(sc * (float)plane[t.idx[1]] + sh, 0.f) +
              t.w[2] * fmaxf(sc * (float)plane[t.idx[2]] + sh, 0.f) +
              t.w[3] * fmaxf(sc * (float)plane[t.idx[3]] + sh, 0.f);
        } else {
          v = t.w[0] * (float)plane[t.idx[0]] + t.w[1] * (float)plane[t.idx[1]] +
              t.w[2] * (float)plane[t.idx[2]] + t.w[3] * (float)plane[t.idx[3]];
        }
      }
      dst[row * LDA + bcol] = (bf16_t)v;
    }
  };

  build(0, sA[0]);
  for (int kt = 0; kt < KT; ++kt) {
    __syncthreads();                                 // tile kt ready; buffer kt-1 readers done
    if (kt + 1 < KT) build(kt + 1, sA[(kt + 1) & 1]);  // overlap next gather with WMMA below
    const bf16_t* sbuf = sA[kt & 1];

    v16bf bf[TN];
#pragma unroll
    for (int tn = 0; tn < TN; ++tn) {
      const bf16_t* p = wt + (long)(wnBase + tn * 16 + bnc) * KP + kt * 32 + bkb;
      bf[tn] = cat16(*(const v8bf*)p, *(const v8bf*)(p + 16));
    }
#pragma unroll
    for (int tm = 0; tm < TM; ++tm) {
      const bf16_t* p = &sbuf[(wmBase + tm * 16 + arow) * LDA + akb];
      v16bf af = cat16(*(const v8bf*)p, *(const v8bf*)(p + 16));
#pragma unroll
      for (int tn = 0; tn < TN; ++tn)
        acc[tm][tn] = __builtin_amdgcn_wmma_f32_16x16x32_bf16(
            false, af, false, bf[tn], (short)0, acc[tm][tn], false, false);
    }
  }

  // ---- epilogue 1: raw conv -> NCHW bf16; 8 consecutive M per lane => one b128 store ----
  const int mhalf = (lane >> 4) * 8;
  const int nloc  = lane & 15;
#pragma unroll
  for (int tm = 0; tm < TM; ++tm)
#pragma unroll
    for (int tn = 0; tn < TN; ++tn) {
      long m0 = mBase + wmBase + tm * 16 + mhalf;
      int n = wnBase + tn * 16 + nloc;
      int b = (int)(m0 >> LOG_HW);
      int s0 = (int)(m0 & (HW - 1));
      v8bf pk;
#pragma unroll
      for (int r = 0; r < 8; ++r) pk[r] = (bf16_t)acc[tm][tn][r];
      *(v8bf*)(out + ((long)b * COUT + n) * HW + s0) = pk;
    }

  // ---- epilogue 2: deterministic per-block BN partials from fp32 accumulators ----
#pragma unroll
  for (int tn = 0; tn < TN; ++tn) {
    float s1 = 0.f, s2 = 0.f;
#pragma unroll
    for (int tm = 0; tm < TM; ++tm)
#pragma unroll
      for (int r = 0; r < 8; ++r) { float v = acc[tm][tn][r]; s1 += v; s2 += v * v; }
    s1 += __shfl_xor(s1, 16, 32);  // combine the two M half-lane groups
    s2 += __shfl_xor(s2, 16, 32);
    if (lane < 16) {
      sRed[0][waveId][wnBase + tn * 16 + lane] = s1;
      sRed[1][waveId][wnBase + tn * 16 + lane] = s2;
    }
  }
  __syncthreads();
  if (tid < BN) {
    int wn2 = tid / (TN * 16);
    float t1 = 0.f, t2 = 0.f;
#pragma unroll
    for (int w2 = 0; w2 < WGM; ++w2) {
      t1 += sRed[0][wn2 * WGM + w2][tid];
      t2 += sRed[1][wn2 * WGM + w2][tid];
    }
    ps1[(long)tid * gridDim.x + blockIdx.x] = t1;
    ps2[(long)tid * gridDim.x + blockIdx.x] = t2;
  }
}

// ---------------- fold per-block partials -> per-channel scale/shift ----------------
__global__ void bn_finalize(const float* __restrict__ ps1, const float* __restrict__ ps2,
                            float* __restrict__ scsh, const float* __restrict__ gamma,
                            const float* __restrict__ beta, int C, int NB, float invN) {
  int c = blockIdx.x * blockDim.x + threadIdx.x;
  if (c >= C) return;
  float s1 = 0.f, s2 = 0.f;
  for (int j = 0; j < NB; ++j) { s1 += ps1[(long)c * NB + j]; s2 += ps2[(long)c * NB + j]; }
  float mean = s1 * invN;
  float var  = s2 * invN - mean * mean;
  float rs = rsqrtf(var + 1e-5f);
  float sc = gamma[c] * rs;
  scsh[c] = sc;
  scsh[C + c] = beta[c] - mean * sc;
}

// ---------------- BN+ReLU + maxpool 2x2 (reads raw conv, writes activated pooled) ----------------
__global__ void maxpool_bn(const bf16_t* __restrict__ in, bf16_t* __restrict__ out,
                           const float* __restrict__ scsh, int C, int H, int W, long total) {
  long i = (long)blockIdx.x * blockDim.x + threadIdx.x;
  if (i >= total) return;
  int OW = W >> 1, OH = H >> 1;
  int ow = (int)(i % OW); long t = i / OW;
  int oh = (int)(t % OH); t /= OH;
  int c = (int)(t % C);   int b = (int)(t / C);
  float sc = scsh[c], sh = scsh[C + c];
  const bf16_t* p = in + (((long)b * C + c) * H + oh * 2) * W + ow * 2;
  float a0 = fmaxf(sc * (float)p[0] + sh, 0.f);
  float a1 = fmaxf(sc * (float)p[1] + sh, 0.f);
  float a2 = fmaxf(sc * (float)p[W] + sh, 0.f);
  float a3 = fmaxf(sc * (float)p[W + 1] + sh, 0.f);
  out[i] = (bf16_t)fmaxf(fmaxf(a0, a1), fmaxf(a2, a3));
}

// ---------------- BN+ReLU + global avg pool 8x8 -> [B,128] fp32 ----------------
__global__ void avgpool_bn(const bf16_t* __restrict__ in, float* __restrict__ out,
                           const float* __restrict__ scsh, long total) {
  long i = (long)blockIdx.x * blockDim.x + threadIdx.x;
  if (i >= total) return;
  int c = (int)(i & 127);
  float sc = scsh[c], sh = scsh[128 + c];
  const bf16_t* p = in + i * 64;
  float s = 0.f;
#pragma unroll
  for (int j = 0; j < 64; ++j) s += fmaxf(sc * (float)p[j] + sh, 0.f);
  out[i] = s * (1.f / 64.f);
}

// ---------------- FC [B,128] x [10,128]^T + b ----------------
__global__ void fck(const float* __restrict__ pooled, const float* __restrict__ w,
                    const float* __restrict__ bias, float* __restrict__ out) {
  int i = blockIdx.x * blockDim.x + threadIdx.x;
  if (i >= 1024 * 10) return;
  int b = i / 10, j = i - b * 10;
  float s = bias[j];
#pragma unroll 8
  for (int c = 0; c < 128; ++c) s += pooled[b * 128 + c] * w[j * 128 + c];
  out[i] = s;
}

extern "C" void kernel_launch(void* const* d_in, const int* in_sizes, int n_in,
                              void* d_out, int out_size, void* d_ws, size_t ws_size,
                              hipStream_t stream) {
  (void)in_sizes; (void)n_in; (void)out_size; (void)ws_size;
  const float* x   = (const float*)d_in[0];
  const float* w11 = (const float*)d_in[1];
  const float* g11 = (const float*)d_in[2];
  const float* b11 = (const float*)d_in[3];
  const float* w12 = (const float*)d_in[4];
  const float* g12 = (const float*)d_in[5];
  const float* b12 = (const float*)d_in[6];
  const float* w21 = (const float*)d_in[7];
  const float* g21 = (const float*)d_in[8];
  const float* b21 = (const float*)d_in[9];
  const float* w22 = (const float*)d_in[10];
  const float* g22 = (const float*)d_in[11];
  const float* b22 = (const float*)d_in[12];
  const float* w31 = (const float*)d_in[13];
  const float* g31 = (const float*)d_in[14];
  const float* b31 = (const float*)d_in[15];
  const float* w32 = (const float*)d_in[16];
  const float* g32 = (const float*)d_in[17];
  const float* b32 = (const float*)d_in[18];
  const float* fcw = (const float*)d_in[19];
  const float* fcb = (const float*)d_in[20];
  const float* c32 = (const float*)d_in[21];
  const float* c16 = (const float*)d_in[22];
  const float* c8  = (const float*)d_in[23];
  float* out = (float*)d_out;

  char* ws = (char*)d_ws;
  size_t off = 0;
  auto take = [&](size_t bytes) -> char* {
    char* p = ws + off;
    off += (bytes + 255) & ~(size_t)255;
    return p;
  };
  bf16_t* actA   = (bf16_t*)take(67108864);          // [1024,32,32,32] bf16 max
  bf16_t* actB   = (bf16_t*)take(67108864);
  bf16_t* wt     = (bf16_t*)take(128 * 1152 * 2);    // transposed bf16 weights (per-layer reuse)
  Tap*    taps32 = (Tap*)take(32 * 32 * 9 * sizeof(Tap));
  Tap*    taps16 = (Tap*)take(16 * 16 * 9 * sizeof(Tap));
  Tap*    taps8  = (Tap*)take(8 * 8 * 9 * sizeof(Tap));
  float*  ps1    = (float*)take(128 * 8192 * 4);     // per-block BN partials (max C x max blocks)
  float*  ps2    = (float*)take(128 * 8192 * 4);
  float*  scsh   = (float*)take(256 * 4);            // current layer scale/shift
  float*  pooled = (float*)take(1024 * 128 * 4);

  tapprep<<<(32 * 32 * 9 + 255) / 256, 256, 0, stream>>>(c32, taps32, 32, 32);
  tapprep<<<(16 * 16 * 9 + 255) / 256, 256, 0, stream>>>(c16, taps16, 16, 16);
  tapprep<<<(8 * 8 * 9 + 255) / 256,   256, 0, stream>>>(c8,  taps8,  8,  8);
  f2bf<<<(1024 * 1024) / 256, 256, 0, stream>>>(x, actA, (long)1024 * 1024);

  // ---- Layer 1.1: 1 -> 32 @ 32x32 (input already "activated") ----
  wprep<<<(32 * 32 + 255) / 256, 256, 0, stream>>>(w11, wt, 32, 9, 32);
  dconv_gemm<1, 32, 32, 32, 8, 1, false><<<8192, 256, 0, stream>>>(actA, wt, taps32, scsh, actB, ps1, ps2);
  bn_finalize<<<1, 128, 0, stream>>>(ps1, ps2, scsh, g11, b11, 32, 8192, 1.f / 1048576.f);

  // ---- Layer 1.2: 32 -> 32 @ 32x32 (applies bn11 while gathering) ----
  wprep<<<(32 * 288 + 255) / 256, 256, 0, stream>>>(w12, wt, 32, 288, 288);
  dconv_gemm<32, 32, 32, 32, 8, 1, true><<<8192, 256, 0, stream>>>(actB, wt, taps32, scsh, actA, ps1, ps2);
  bn_finalize<<<1, 128, 0, stream>>>(ps1, ps2, scsh, g12, b12, 32, 8192, 1.f / 1048576.f);
  maxpool_bn<<<8388608 / 256, 256, 0, stream>>>(actA, actB, scsh, 32, 32, 32, 8388608L);

  // ---- Layer 2.1: 32 -> 64 @ 16x16 (pooled input already activated) ----
  wprep<<<(64 * 288 + 255) / 256, 256, 0, stream>>>(w21, wt, 64, 288, 288);
  dconv_gemm<32, 64, 16, 16, 4, 2, false><<<2048, 256, 0, stream>>>(actB, wt, taps16, scsh, actA, ps1, ps2);
  bn_finalize<<<1, 128, 0, stream>>>(ps1, ps2, scsh, g21, b21, 64, 2048, 1.f / 262144.f);

  // ---- Layer 2.2: 64 -> 64 @ 16x16 (applies bn21) ----
  wprep<<<(64 * 576 + 255) / 256, 256, 0, stream>>>(w22, wt, 64, 576, 576);
  dconv_gemm<64, 64, 16, 16, 4, 2, true><<<2048, 256, 0, stream>>>(actA, wt, taps16, scsh, actB, ps1, ps2);
  bn_finalize<<<1, 128, 0, stream>>>(ps1, ps2, scsh, g22, b22, 64, 2048, 1.f / 262144.f);
  maxpool_bn<<<4194304 / 256, 256, 0, stream>>>(actB, actA, scsh, 64, 16, 16, 4194304L);

  // ---- Layer 3.1: 64 -> 128 @ 8x8 ----
  wprep<<<(128 * 576 + 255) / 256, 256, 0, stream>>>(w31, wt, 128, 576, 576);
  dconv_gemm<64, 128, 8, 8, 2, 4, false><<<512, 256, 0, stream>>>(actA, wt, taps8, scsh, actB, ps1, ps2);
  bn_finalize<<<1, 128, 0, stream>>>(ps1, ps2, scsh, g31, b31, 128, 512, 1.f / 65536.f);

  // ---- Layer 3.2: 128 -> 128 @ 8x8 (applies bn31) ----
  wprep<<<(128 * 1152 + 255) / 256, 256, 0, stream>>>(w32, wt, 128, 1152, 1152);
  dconv_gemm<128, 128, 8, 8, 2, 4, true><<<512, 256, 0, stream>>>(actB, wt, taps8, scsh, actA, ps1, ps2);
  bn_finalize<<<1, 128, 0, stream>>>(ps1, ps2, scsh, g32, b32, 128, 512, 1.f / 65536.f);

  // ---- head: bn32+ReLU fused into avg pool, then tiny FC ----
  avgpool_bn<<<131072 / 256, 256, 0, stream>>>(actA, pooled, scsh, 131072L);
  fck<<<(10240 + 255) / 256, 256, 0, stream>>>(pooled, fcw, fcb, out);
}
